// MultiHeadAttention_7352984010953
// MI455X (gfx1250) — matrix operand order repaired and checked by reading, not compiled, not run on hardware
//
#include <hip/hip_runtime.h>

// ---------------------------------------------------------------------------
// MI455X (gfx1250, CDNA5, wave32) fused multi-head attention forward.
// Matrix math: V_WMMA_F32_16X16X32_F16. K/V staging: Tensor Data Mover
// (tensor_load_to_lds + s_wait_tensorcnt). Softmax: permlane16 row-max +
// ones-column WMMA denominator.
// ---------------------------------------------------------------------------

#define DEVINL __device__ __forceinline__

typedef __attribute__((ext_vector_type(16))) _Float16 v16h;
typedef __attribute__((ext_vector_type(8)))  _Float16 v8h;
typedef __attribute__((ext_vector_type(8)))  float    v8f;
typedef __attribute__((ext_vector_type(4)))  float    v4f;

typedef unsigned int u32x4 __attribute__((ext_vector_type(4)));
typedef int          i32x4 __attribute__((ext_vector_type(4)));
typedef int          i32x8 __attribute__((ext_vector_type(8)));

constexpr int BATCH = 2;
constexpr int SEQ   = 2048;
constexpr int DM    = 1024;
constexpr int NH    = 16;
constexpr int DH    = 64;

// ---------------- fragment helpers ----------------

DEVINL v16h cat8(v8h lo, v8h hi) {
  return __builtin_shufflevector(lo, hi, 0,1,2,3,4,5,6,7,8,9,10,11,12,13,14,15);
}

// 8 consecutive fp32 -> 8 f16
DEVINL v8h cvt8(const float* p) {
  v4f a = *(const v4f*)p;
  v4f b = *(const v4f*)(p + 4);
  v8h r;
#pragma unroll
  for (int i = 0; i < 4; ++i) { r[i] = (_Float16)a[i]; r[i+4] = (_Float16)b[i]; }
  return r;
}

// A fragment (16x32): lane = row, K runs {half*8, 16+half*8}; row length 32
DEVINL v16h load_a_row32(const _Float16* row, int half) {
  v8h lo = *(const v8h*)(row + half*8);
  v8h hi = *(const v8h*)(row + 16 + half*8);
  return cat8(lo, hi);
}
// A fragment from a longer row with chunk base k0
DEVINL v16h load_a_f16(const _Float16* row, int k0, int half) {
  v8h lo = *(const v8h*)(row + k0 + half*8);
  v8h hi = *(const v8h*)(row + k0 + 16 + half*8);
  return cat8(lo, hi);
}
// B fragment (32x16): lane = column, K runs {base+half*8, base+16+half*8}; koff = base + half*16
DEVINL v16h load_b_run(const _Float16* row, int koff) {
  const int base = koff & ~31, hf = (koff >> 4) & 1;
  v8h lo = *(const v8h*)(row + base + hf * 8);
  v8h hi = *(const v8h*)(row + base + 16 + hf * 8);
  return cat8(lo, hi);
}

DEVINL v8f wmma16(v16h a, v16h b, v8f c) {
  return __builtin_amdgcn_wmma_f32_16x16x32_f16(false, a, false, b, (short)0, c,
                                                false, false);
}

// ---- row-max reduction across a 16-lane group (wave32 halves independent) ----
#if __has_builtin(__builtin_amdgcn_permlane16)
DEVINL float permmax(float x, unsigned s0, unsigned s1) {
  int y = __builtin_amdgcn_permlane16(__float_as_int(x), __float_as_int(x),
                                      s0, s1, false, false);
  return fmaxf(x, __int_as_float(y));
}
DEVINL float redmax16(float x) {
  x = permmax(x, 0x67452301u, 0xEFCDAB89u);   // xor 1
  x = permmax(x, 0x54761032u, 0xDCFE98BAu);   // xor 2
  x = permmax(x, 0x32107654u, 0xBA98FEDCu);   // xor 4
  x = permmax(x, 0xFEDCBA98u, 0x76543210u);   // xor 8
  return x;
}
#else
DEVINL float redmax16(float x) {
#pragma unroll
  for (int m = 1; m < 16; m <<= 1) x = fmaxf(x, __shfl_xor(x, m, 32));
  return x;
}
#endif

// ---- Tensor Data Mover: 2D f16 tile load, global -> LDS (ISA ch.8 D#) ----
// Loads tile_d1 rows of tile_d0 contiguous f16 elements; row r starts at
// gaddr + r*stride_d0 elements; rows land compacted in LDS at lds_addr.
DEVINL void tdm_load_2d(unsigned lds_addr, const void* gaddr,
                        unsigned tensor_d0, unsigned tensor_d1,
                        unsigned tile_d0, unsigned tile_d1,
                        unsigned stride_d0) {
  const unsigned long long ga = (unsigned long long)gaddr;
  u32x4 g0;
  g0[0] = 1u;                                          // count=1, user mode
  g0[1] = lds_addr;                                    // LDS byte address
  g0[2] = (unsigned)(ga & 0xFFFFFFFFu);                // global_addr[31:0]
  g0[3] = (unsigned)((ga >> 32) & 0x01FFFFFFu) | (2u << 30);  // addr[56:32]|type=2
  i32x8 g1;
  g1[0] = 0x00010000;                                  // data_size = 2 bytes
  g1[1] = (int)((tensor_d0 & 0xFFFFu) << 16);          // tensor_dim0 lo16
  g1[2] = (int)((tensor_d0 >> 16) | ((tensor_d1 & 0xFFFFu) << 16));
  g1[3] = (int)((tensor_d1 >> 16) | (tile_d0 << 16));  // tile_dim0
  g1[4] = (int)tile_d1;                                // tile_dim1 (tile_dim2=0)
  g1[5] = (int)stride_d0;                              // tensor_dim0_stride lo32
  g1[6] = 0;
  g1[7] = 0;
  const i32x4 gz = {0, 0, 0, 0};
#if defined(__clang_major__) && (__clang_major__ >= 23)
  const i32x8 gz8 = {0, 0, 0, 0, 0, 0, 0, 0};
  __builtin_amdgcn_tensor_load_to_lds(g0, g1, gz, gz, gz8, 0);
#else
  __builtin_amdgcn_tensor_load_to_lds(g0, g1, gz, gz, 0);
#endif
}

// ---------------- projection GEMM:  out = X @ W^T + b ----------------
// Block: 256 threads (8 waves), 128x128 output tile, K-step 32.
// A/B tiles staged fp32->f16 in LDS, double-buffered, 1 barrier per step.

enum { MODE_QK = 0, MODE_V = 1, MODE_O = 2 };

template <int MODE>
__global__ __launch_bounds__(256)
void proj_gemm(const float* __restrict__ X, const float* __restrict__ W,
               const float* __restrict__ bias,
               _Float16* __restrict__ outh, float* __restrict__ outf) {
  __shared__ _Float16 Atile[2][128 * 32];   // 8 KB each buffer
  __shared__ _Float16 Btile[2][128 * 32];

  const int tid  = threadIdx.x;
  const int lane = tid & 31;
  const int wv   = tid >> 5;                // 0..7
  const int wm   = wv >> 1;                 // 0..3 -> 32-row slice
  const int wn   = wv & 1;                  // 0..1 -> 64-col slice
  const int mb   = blockIdx.x >> 3;         // 0..31
  const int nb   = blockIdx.x & 7;          // 0..7
  const int rowB = mb * 128;
  const int colB = nb * 128;
  const int rA   = lane & 15;
  const int half = lane >> 4;

  // cooperative copy mapping: thread -> (row, 16-float chunk)
  const int cr = tid >> 1;                  // 0..127
  const int ck = (tid & 1) * 16;            // 0 or 16
  const float* aSrc = X + (size_t)(rowB + cr) * DM + ck;
  const float* bSrc = W + (size_t)(colB + cr) * DM + ck;

  // stage k-step 0
  {
    v8h a0 = cvt8(aSrc), a1 = cvt8(aSrc + 8);
    v8h b0 = cvt8(bSrc), b1 = cvt8(bSrc + 8);
    *(v8h*)&Atile[0][cr * 32 + ck]     = a0;
    *(v8h*)&Atile[0][cr * 32 + ck + 8] = a1;
    *(v8h*)&Btile[0][cr * 32 + ck]     = b0;
    *(v8h*)&Btile[0][cr * 32 + ck + 8] = b1;
  }
  __syncthreads();

  v8f z = {};
  v8f acc[2][4];
#pragma unroll
  for (int i = 0; i < 2; ++i)
#pragma unroll
    for (int t = 0; t < 4; ++t) acc[i][t] = z;

  for (int ks = 0; ks < DM / 32; ++ks) {
    const int cur = ks & 1, nxt = cur ^ 1;
    const bool more = (ks + 1) < DM / 32;

    // issue next-step global loads early (hidden under WMMAs)
    v8h na0, na1, nb0, nb1;
    if (more) {
      const float* ap = aSrc + (ks + 1) * 32;
      const float* bp = bSrc + (ks + 1) * 32;
      na0 = cvt8(ap); na1 = cvt8(ap + 8);
      nb0 = cvt8(bp); nb1 = cvt8(bp + 8);
    }

    const _Float16* At = &Atile[cur][0];
    const _Float16* Bt = &Btile[cur][0];
    v16h a0 = load_a_row32(At + (wm * 32 +      rA) * 32, half);
    v16h a1 = load_a_row32(At + (wm * 32 + 16 + rA) * 32, half);
#pragma unroll
    for (int t = 0; t < 4; ++t) {
      v16h b = load_b_run(Bt + (wn * 64 + t * 16 + rA) * 32, half * 16);
      acc[0][t] = wmma16(a0, b, acc[0][t]);
      acc[1][t] = wmma16(a1, b, acc[1][t]);
    }

    if (more) {
      *(v8h*)&Atile[nxt][cr * 32 + ck]     = na0;
      *(v8h*)&Atile[nxt][cr * 32 + ck + 8] = na1;
      *(v8h*)&Btile[nxt][cr * 32 + ck]     = nb0;
      *(v8h*)&Btile[nxt][cr * 32 + ck + 8] = nb1;
    }
    __syncthreads();
  }

  // epilogue: C/D layout row = v + 8*half, col = lane&15
#pragma unroll
  for (int i = 0; i < 2; ++i) {
#pragma unroll
    for (int t = 0; t < 4; ++t) {
      const int col = colB + wn * 64 + t * 16 + rA;
      const float bv = bias[col];
#pragma unroll
      for (int v = 0; v < 8; ++v) {
        const int row = rowB + wm * 32 + i * 16 + v + 8 * half;
        const float val = acc[i][t][v] + bv;
        if (MODE == MODE_O) {
          outf[(size_t)row * DM + col] = val;
        } else {
          const int bb = row >> 11;
          const int ss = row & (SEQ - 1);
          const int hh = col >> 6;
          const int dd = col & (DH - 1);
          size_t idx;
          if (MODE == MODE_QK)
            idx = (((size_t)(bb * NH + hh) * SEQ) + ss) * DH + dd;  // [B,H,S,Dh]
          else
            idx = (((size_t)(bb * NH + hh) * DH) + dd) * SEQ + ss;  // [B,H,Dh,S]
          outh[idx] = (_Float16)val;
        }
      }
    }
  }
}

// ---------------- flash attention ----------------
// Block: 256 threads = 8 waves = 8 query tiles (16 rows each) of ONE (b,h).
// K/V tiles (32 keys) DMA'd into LDS by the Tensor Data Mover (wave 0 issues,
// TENSORcnt tracked), double-buffered, shared by all 8 waves.
// Softmax denominator via ones-column WMMA accumulator (no sum reduction).

__global__ __launch_bounds__(256)
void flash_attn(const _Float16* __restrict__ Qh, const _Float16* __restrict__ Kh,
                const _Float16* __restrict__ Vt, float* __restrict__ ctx) {
  __shared__ _Float16 Ktile[2][32 * 64];    // 4 KB each
  __shared__ _Float16 Vtile[2][64 * 32];    // 4 KB each
  __shared__ _Float16 pbuf[8][16 * 32];     // per-wave P transpose staging

  const int tid  = threadIdx.x;
  const int lane = tid & 31;
  const int wv   = tid >> 5;
  const int qt   = blockIdx.x * 8 + wv;     // 0..4095 query tiles
  const int bh   = qt >> 7;                 // all 8 waves share (b,h)
  const int q0   = (qt & 127) * 16;
  const int b    = bh >> 4;
  const int h    = bh & 15;
  const int rA   = lane & 15;
  const int half = lane >> 4;

  const _Float16* Qb = Qh + (size_t)bh * SEQ * DH;
  const _Float16* Kb = Kh + (size_t)bh * SEQ * DH;
  const _Float16* Vb = Vt + (size_t)bh * DH  * SEQ;

  const unsigned ldsK[2] = { (unsigned)(uintptr_t)&Ktile[0][0],
                             (unsigned)(uintptr_t)&Ktile[1][0] };
  const unsigned ldsV[2] = { (unsigned)(uintptr_t)&Vtile[0][0],
                             (unsigned)(uintptr_t)&Vtile[1][0] };

  const _Float16* qrow = Qb + (size_t)(q0 + rA) * DH;
  const v16h qf0 = load_a_f16(qrow,  0, half);
  const v16h qf1 = load_a_f16(qrow, 32, half);

  v16h ones;
#pragma unroll
  for (int i = 0; i < 16; ++i) ones[i] = (_Float16)1.0f;

  v8f z = {};
  v8f o[4], oS = z;                         // oS accumulates P @ 1 (row sums)
#pragma unroll
  for (int t = 0; t < 4; ++t) o[t] = z;
  float m8[8];
#pragma unroll
  for (int v = 0; v < 8; ++v) m8[v] = -3.0e38f;

  // TDM-stage key tile 0 (wave 0 issues; TENSORcnt; publish via barrier)
  if (wv == 0) {
    tdm_load_2d(ldsK[0], Kb, DH, SEQ, DH, 32, DH);      // 32 rows x 64 contig
    tdm_load_2d(ldsV[0], Vb, SEQ, DH, 32, DH, SEQ);     // 64 rows x 32, stride S
    __builtin_amdgcn_s_wait_tensorcnt(0);
  }
  __syncthreads();

  for (int kb = 0; kb < SEQ / 32; ++kb) {
    const int key0 = kb * 32;
    const int cur = kb & 1, nxt = cur ^ 1;
    const bool more = (kb + 1) < SEQ / 32;

    // kick off next tile's DMA immediately; it overlaps all compute below
    if (more && wv == 0) {
      tdm_load_2d(ldsK[nxt], Kb + (size_t)(key0 + 32) * DH, DH, SEQ, DH, 32, DH);
      tdm_load_2d(ldsV[nxt], Vb + key0 + 32,                SEQ, DH, 32, DH, SEQ);
    }

    const _Float16* Kt = Ktile[cur];
    const _Float16* Vw = Vtile[cur];

    // scores: two 16x16 tiles over 32 keys (K-dim = Dh = 64 -> 2 WMMA each)
    const _Float16* kr0 = Kt + (     rA) * 64;
    const _Float16* kr1 = Kt + (16 + rA) * 64;
    v16h k00 = load_b_run(kr0,      half * 16);
    v16h k01 = load_b_run(kr0, 32 + half * 16);
    v16h k10 = load_b_run(kr1,      half * 16);
    v16h k11 = load_b_run(kr1, 32 + half * 16);
    v8f s0 = wmma16(qf1, k01, wmma16(qf0, k00, z)) * 0.125f;
    v8f s1 = wmma16(qf1, k11, wmma16(qf0, k10, z)) * 0.125f;

    // online softmax: row-max (permlane16 butterfly) + rescale
    float corr[8];
#pragma unroll
    for (int v = 0; v < 8; ++v) {
      float tv = redmax16(fmaxf(s0[v], s1[v]));
      float nm = fmaxf(m8[v], tv);
      corr[v]  = __expf(m8[v] - nm);
      m8[v]    = nm;
    }
#pragma unroll
    for (int v = 0; v < 8; ++v) {
      s0[v] = __expf(s0[v] - m8[v]);
      s1[v] = __expf(s1[v] - m8[v]);
    }
#pragma unroll
    for (int t = 0; t < 4; ++t)
#pragma unroll
      for (int v = 0; v < 8; ++v) o[t][v] *= corr[v];
#pragma unroll
    for (int v = 0; v < 8; ++v) oS[v] *= corr[v];

    // transpose P (16q x 32k): C/D layout -> A layout through LDS
#pragma unroll
    for (int v = 0; v < 8; ++v) {
      pbuf[wv][(v + 8 * half) * 32 +      rA] = (_Float16)s0[v];
      pbuf[wv][(v + 8 * half) * 32 + 16 + rA] = (_Float16)s1[v];
    }
    asm volatile("s_wait_dscnt 0x0" ::: "memory");
    const v16h pf = load_a_row32(&pbuf[wv][rA * 32], half);

    // O += P @ V ;  oS += P @ 1
#pragma unroll
    for (int t = 0; t < 4; ++t) {
      v16h vf = load_b_run(Vw + (t * 16 + rA) * 32, half * 16);
      o[t] = wmma16(pf, vf, o[t]);
    }
    oS = wmma16(pf, ones, oS);

    // wave 0 makes sure the DMA for the next buffer has landed, then publish
    if (more && wv == 0) __builtin_amdgcn_s_wait_tensorcnt(0);
    __syncthreads();
  }

  float inv[8];
#pragma unroll
  for (int v = 0; v < 8; ++v) inv[v] = 1.0f / oS[v];
#pragma unroll
  for (int t = 0; t < 4; ++t)
#pragma unroll
    for (int v = 0; v < 8; ++v) {
      const int srow = q0 + v + 8 * half;
      ctx[((size_t)b * SEQ + srow) * DM + h * DH + t * 16 + rA] = o[t][v] * inv[v];
    }
}

// ---------------- host launcher ----------------

extern "C" void kernel_launch(void* const* d_in, const int* in_sizes, int n_in,
                              void* d_out, int out_size, void* d_ws, size_t ws_size,
                              hipStream_t stream) {
  (void)in_sizes; (void)n_in; (void)out_size; (void)ws_size;
  const float* q  = (const float*)d_in[0];
  const float* k  = (const float*)d_in[1];
  const float* v  = (const float*)d_in[2];
  const float* Wq = (const float*)d_in[3];
  const float* bq = (const float*)d_in[4];
  const float* Wk = (const float*)d_in[5];
  const float* bk = (const float*)d_in[6];
  const float* Wv = (const float*)d_in[7];
  const float* bv = (const float*)d_in[8];
  const float* Wo = (const float*)d_in[9];
  const float* bo = (const float*)d_in[10];
  float* out = (float*)d_out;

  // workspace: Qh(8MB) Kh(8MB) Vt(8MB) ctx fp32(16MB) = 40MB
  char* ws = (char*)d_ws;
  constexpr size_t SZH = (size_t)BATCH * NH * SEQ * DH * sizeof(_Float16);
  _Float16* Qh  = (_Float16*)(ws);
  _Float16* Kh  = (_Float16*)(ws + SZH);
  _Float16* Vt  = (_Float16*)(ws + 2 * SZH);
  float*    ctx = (float*)   (ws + 3 * SZH);

  const dim3 blk(256);                              // 8 waves (wave32)
  const int gemm_blocks = (BATCH * SEQ / 128) * (DM / 128);  // 256
  const int attn_blocks = (BATCH * NH * SEQ / 16) / 8;       // 512

  proj_gemm<MODE_QK><<<gemm_blocks, blk, 0, stream>>>(q, Wq, bq, Qh, nullptr);
  proj_gemm<MODE_QK><<<gemm_blocks, blk, 0, stream>>>(k, Wk, bk, Kh, nullptr);
  proj_gemm<MODE_V ><<<gemm_blocks, blk, 0, stream>>>(v, Wv, bv, Vt, nullptr);
  flash_attn<<<attn_blocks, blk, 0, stream>>>(Qh, Kh, Vt, ctx);
  proj_gemm<MODE_O ><<<gemm_blocks, blk, 0, stream>>>(ctx, Wo, bo, nullptr, out);
}
